// BiDAF_59992103190921
// MI455X (gfx1250) — matrix operand order repaired and checked by reading, not compiled, not run on hardware
//
#include <hip/hip_runtime.h>
#include <hip/hip_bf16.h>

// ---------------------------------------------------------------------------
// Types for CDNA5 WMMA (gfx1250, wave32): D(f32 16x16) = A(bf16 16x32) x B(bf16 32x16) + C
// ---------------------------------------------------------------------------
typedef __bf16 bf16;
typedef __bf16 v16bf __attribute__((ext_vector_type(16)));
typedef float  v8f   __attribute__((ext_vector_type(8)));

__device__ __forceinline__ v8f vzero8() {
  v8f z;
#pragma unroll
  for (int e = 0; e < 8; e++) z[e] = 0.0f;
  return z;
}
// A-fragment: lane half h: k = 16h..16h+15 (contiguous, matches B)
__device__ __forceinline__ v16bf load_fragA(const bf16* row, int half) {
  v16bf f;
  const bf16* p = row + 16 * half;
#pragma unroll
  for (int e = 0; e < 16; e++) f[e] = p[e];
  return f;
}
// B-fragment: lanes 0-15 hold k=0..15, lanes 16-31 hold k=16..31 (contiguous)
__device__ __forceinline__ v16bf load_fragB(const bf16* row, int half) {
  v16bf f;
  const bf16* p = row + 16 * half;
#pragma unroll
  for (int e = 0; e < 16; e++) f[e] = p[e];
  return f;
}
__device__ __forceinline__ v8f wmma_bf16(v16bf a, v16bf b, v8f c) {
  return __builtin_amdgcn_wmma_f32_16x16x32_bf16(false, a, false, b, (short)0, c,
                                                 false, false);
}
__device__ __forceinline__ float sigm(float x) { return 1.0f / (1.0f + __expf(-x)); }

// ---- CDNA5 async memory->LDS copy (ASYNCcnt path, no VGPR round-trip) ----
// 32-bit LDS offset of a generic pointer known to point into LDS
__device__ __forceinline__ unsigned lds_off(const void* p) {
  return (unsigned)(unsigned long long)(__attribute__((address_space(3))) const char*)p;
}
// copy 32 bytes global->LDS: INST_OFFSET applies to BOTH addresses (ISA 10.x)
__device__ __forceinline__ void async_copy32(unsigned lds, const bf16* g) {
  asm volatile(
      "global_load_async_to_lds_b128 %0, %1, off\n\t"
      "global_load_async_to_lds_b128 %0, %1, off offset:16"
      :: "v"(lds), "v"(g) : "memory");
}
__device__ __forceinline__ void wait_async0() {
  asm volatile("s_wait_asynccnt 0x0" ::: "memory");
}

// ---------------------------------------------------------------------------
// Model constants
// ---------------------------------------------------------------------------
#define BB   32
#define CLEN 800
#define QLEN 48
#define NCR  (BB * CLEN)   // 25600 context rows (multiple of 64)
#define NQR  (BB * QLEN)   // 1536 query rows (multiple of 64)
#define DD2  400
#define HH   200
#define GG4  800
#define GG8  1600
#define KPAD 224           // 200 -> 7*32  (recurrent GEMM K)
#define KD2P 416           // 400 -> 13*32 (feature GEMM K)

// ---------------------------------------------------------------------------
// WMMA GEMM:  Y[N x M] = X[N x Kp] * W[M x Kp]^T (+bias0 +bias1) (+act)
// Operands pre-converted bf16, K zero-padded to 32. N multiple of 64.
// Double-buffered LDS filled by global_load_async_to_lds_b128: the async
// engine streams chunk k+1 while the waves run WMMA on chunk k.
// act: 0=none 1=relu 2=sigmoid
// ---------------------------------------------------------------------------
__global__ __launch_bounds__(128) void gemm_wmma(
    const bf16* __restrict__ X, const bf16* __restrict__ W,
    const float* __restrict__ bias0, const float* __restrict__ bias1,
    float* __restrict__ Y, int Kp, int M, int ldY, int act) {
  __shared__ __align__(16) bf16 Xs[2][64][40];
  __shared__ __align__(16) bf16 Ws[2][64][40];
  const int tid = threadIdx.x;
  const int lane = tid & 31, wv = tid >> 5;
  const int l16 = lane & 15, half = lane >> 4;
  const int nBase = blockIdx.x * 64;
  const int mBase = blockIdx.y * 64;

  // staging addresses: this thread owns 32 bytes (16 bf16) per tile
  const int row = tid >> 1;
  const int ks = (tid & 1) * 16;
  const bf16* xp = X + (size_t)(nBase + row) * Kp + ks;
  int wrow = mBase + row;
  if (wrow >= M) wrow = M - 1;  // clamp; cols >= M discarded in epilogue
  const bf16* wp = W + (size_t)wrow * Kp + ks;
  const unsigned ldsX0 = lds_off(&Xs[0][row][ks]);
  const unsigned ldsX1 = lds_off(&Xs[1][row][ks]);
  const unsigned ldsW0 = lds_off(&Ws[0][row][ks]);
  const unsigned ldsW1 = lds_off(&Ws[1][row][ks]);

  v8f acc[4];
#pragma unroll
  for (int i = 0; i < 4; i++) acc[i] = vzero8();

  const int nK = Kp >> 5;
  // prologue: chunk 0 -> buffer 0
  async_copy32(ldsX0, xp);
  async_copy32(ldsW0, wp);
  wait_async0();
  __syncthreads();

  for (int kc = 0; kc < nK; kc++) {
    const int cur = kc & 1;
    if (kc + 1 < nK) {  // stream next chunk into the other buffer
      async_copy32(cur ? ldsX0 : ldsX1, xp + (size_t)(kc + 1) * 32);
      async_copy32(cur ? ldsW0 : ldsW1, wp + (size_t)(kc + 1) * 32);
    }
    v16bf a = load_fragA(&Xs[cur][wv * 16 + l16][0], half);
#pragma unroll
    for (int mt = 0; mt < 4; mt++) {
      v16bf b = load_fragB(&Ws[cur][mt * 16 + l16][0], half);
      acc[mt] = wmma_bf16(a, b, acc[mt]);
    }
    if (kc + 1 < nK) {
      wait_async0();     // my async stores to LDS done
      __syncthreads();   // everyone's done -> next buffer valid, old reads done
    }
  }

  const int rbase = half * 8;
#pragma unroll
  for (int mt = 0; mt < 4; mt++) {
    const int col = mBase + mt * 16 + l16;
    if (col >= M) continue;
    float bv = 0.0f;
    if (bias0) bv += bias0[col];
    if (bias1) bv += bias1[col];
#pragma unroll
    for (int r = 0; r < 8; r++) {
      const int orow = nBase + wv * 16 + rbase + r;
      float v = acc[mt][r] + bv;
      if (act == 1) v = fmaxf(v, 0.0f);
      else if (act == 2) v = sigm(v);
      Y[(size_t)orow * ldY + col] = v;
    }
  }
}

// f32 (N x D) -> bf16 (N x ldOut) with zero K-pad; optional elementwise scale
__global__ void bf16_build(const float* __restrict__ X, const float* __restrict__ wv,
                           bf16* __restrict__ out, int N, int D, int ldOut) {
  const size_t i = (size_t)blockIdx.x * blockDim.x + threadIdx.x;
  if (i >= (size_t)N * ldOut) return;
  const int k = (int)(i % ldOut);
  const size_t n = i / ldOut;
  float v = 0.0f;
  if (k < D) {
    v = X[n * D + k];
    if (wv) v *= wv[k];
  }
  out[i] = (bf16)v;
}

// ---------------------------------------------------------------------------
// Char CNN: per word: embed 16 chars x 8, conv (100ch, 8x5) over 12 positions,
// +bias, max-pool -> 100 features.
// ---------------------------------------------------------------------------
__global__ __launch_bounds__(128) void char_conv(
    const int* __restrict__ idx, const float* __restrict__ emb,
    const float* __restrict__ convw, const float* __restrict__ convb,
    float* __restrict__ out) {
  __shared__ float E[16][8];
  const int w = blockIdx.x;
  const int tid = threadIdx.x;
  {
    const int c = tid >> 3, d = tid & 7;
    E[c][d] = emb[(size_t)idx[(size_t)w * 16 + c] * 8 + d];
  }
  __syncthreads();
  if (tid < 100) {
    const float* wp = convw + tid * 40;  // (ch,1,8,5) -> [d*5+x]
    const float bv = convb[tid];
    float m = -1e30f;
    for (int p = 0; p < 12; p++) {
      float s = bv;
#pragma unroll
      for (int d = 0; d < 8; d++)
#pragma unroll
        for (int x = 0; x < 5; x++) s += E[p + x][d] * wp[d * 5 + x];
      m = fmaxf(m, s);
    }
    out[(size_t)w * 100 + tid] = m;
  }
}

// concat [char(100) | word_emb(300)] -> (N,400)
__global__ void concat_cw(const float* __restrict__ ch, const int* __restrict__ wid,
                          const float* __restrict__ wemb, float* __restrict__ out,
                          int N) {
  const size_t i = (size_t)blockIdx.x * blockDim.x + threadIdx.x;
  if (i >= (size_t)N * 400) return;
  const int d = (int)(i % 400);
  const size_t n = i / 400;
  out[i] = (d < 100) ? ch[n * 100 + d] : wemb[(size_t)wid[n] * 300 + (d - 100)];
}

// x = g*h + (1-g)*x   (h relu'd, g sigmoided in gemm epilogue)
__global__ void hw_combine(float* __restrict__ x, const float* __restrict__ h,
                           const float* __restrict__ g, size_t n) {
  const size_t i = (size_t)blockIdx.x * blockDim.x + threadIdx.x;
  if (i < n) x[i] = g[i] * h[i] + (1.0f - g[i]) * x[i];
}

// ---------------------------------------------------------------------------
// Recurrent LSTM direction (persistent, one workgroup per direction).
// h: LDS bf16 (A-fragment layout); z/gates/c in registers of the wave owning
// the WMMA tile. 26 output tiles (2 batch x 13 hidden-col) over 8 waves.
// ---------------------------------------------------------------------------
__global__ __launch_bounds__(256) void lstm_dir(
    const float* __restrict__ xwF, const float* __restrict__ xwB,
    const bf16* __restrict__ whhF, const bf16* __restrict__ whhB,
    float* __restrict__ out, int T, int ldOut) {
  __shared__ __align__(16) bf16 hS[BB][KPAD];
  const int rev = blockIdx.x;  // 0 = forward, 1 = backward
  const float* xw = rev ? xwB : xwF;
  const bf16* whh = rev ? whhB : whhF;
  const int colOff = rev ? HH : 0;
  const int tid = threadIdx.x, lane = tid & 31, wv = tid >> 5;
  const int l16 = lane & 15, half = lane >> 4;

  for (int i = tid; i < BB * KPAD; i += 256) ((bf16*)hS)[i] = (bf16)0.0f;
  __syncthreads();

  int combo[4], nC = 0;
  for (int c = 0; c < 4; c++) {
    const int idx = wv + 8 * c;
    if (idx < 26) combo[nC++] = idx;
  }
  float cst[4][8];
  for (int c = 0; c < 4; c++)
    for (int r = 0; r < 8; r++) cst[c][r] = 0.0f;

  v8f acc[4][4];

  for (int s = 0; s < T; s++) {
    const int t = rev ? (T - 1 - s) : s;
    for (int c = 0; c < nC; c++)
      for (int gg = 0; gg < 4; gg++) acc[c][gg] = vzero8();

    // Phase A: z = h * Whh^T (reads hS only)
    for (int kc = 0; kc < 7; kc++) {
      for (int c = 0; c < nC; c++) {
        const int nt = combo[c] & 1, jt = combo[c] >> 1;
        const int j = jt * 16 + l16;
        const int jc = (j < HH) ? j : (HH - 1);  // clamp: bad lanes unused
        v16bf a = load_fragA(&hS[nt * 16 + l16][kc * 32], half);
#pragma unroll
        for (int gg = 0; gg < 4; gg++) {
          v16bf b = load_fragB(whh + (size_t)(gg * HH + jc) * KPAD + kc * 32, half);
          acc[c][gg] = wmma_bf16(a, b, acc[c][gg]);
        }
      }
    }
    __syncthreads();

    // Phase B: gates + state update in-register; write new h
    for (int c = 0; c < nC; c++) {
      const int nt = combo[c] & 1, jt = combo[c] >> 1;
      const int j = jt * 16 + l16;
      if (j < HH) {
#pragma unroll
        for (int r = 0; r < 8; r++) {
          const int b = nt * 16 + half * 8 + r;
          const size_t base = ((size_t)b * T + t) * GG4;
          const float iv = sigm(acc[c][0][r] + xw[base + j]);
          const float fv = sigm(acc[c][1][r] + xw[base + HH + j]);
          const float gv = tanhf(acc[c][2][r] + xw[base + 2 * HH + j]);
          const float ov = sigm(acc[c][3][r] + xw[base + 3 * HH + j]);
          const float cn = fv * cst[c][r] + iv * gv;
          cst[c][r] = cn;
          const float hv = ov * tanhf(cn);
          hS[b][j] = (bf16)hv;
          out[((size_t)b * T + t) * ldOut + colOff + j] = hv;
        }
      }
    }
    __syncthreads();
  }
}

// row-wise dot:  out[n] = X[n,:] . w + bias[0]
__global__ __launch_bounds__(256) void rowdot(
    const float* __restrict__ X, const float* __restrict__ w,
    const float* __restrict__ bias, float* __restrict__ out, int N, int D) {
  const int row = blockIdx.x * 8 + (threadIdx.x >> 5);
  const int lane = threadIdx.x & 31;
  if (row >= N) return;
  const float* x = X + (size_t)row * D;
  float s = 0.0f;
  for (int d = lane; d < D; d += 32) s += x[d] * w[d];
  for (int off = 16; off; off >>= 1) s += __shfl_down(s, off);
  if (lane == 0) out[row] = s + bias[0];
}

// similarity: s[b,i,j] = (c_i ∘ w_cq) . q_j + sc[b,i] + sq[b,j] + b_cq  (WMMA)
__global__ __launch_bounds__(128) void sim_wmma(
    const bf16* __restrict__ cw, const bf16* __restrict__ qb,
    const float* __restrict__ sc, const float* __restrict__ sq,
    const float* __restrict__ bcq, float* __restrict__ s) {
  const int b = blockIdx.y;
  const int iBase = blockIdx.x * 64;
  const int tid = threadIdx.x, lane = tid & 31, wv = tid >> 5;
  const int l16 = lane & 15, half = lane >> 4;
  v8f acc[3];
#pragma unroll
  for (int jt = 0; jt < 3; jt++) acc[jt] = vzero8();
  int irow = iBase + wv * 16 + l16;
  if (irow >= CLEN) irow = CLEN - 1;
  const bf16* crow = cw + ((size_t)b * CLEN + irow) * KD2P;
  for (int kc = 0; kc < 13; kc++) {
    v16bf a = load_fragA(crow + kc * 32, half);
#pragma unroll
    for (int jt = 0; jt < 3; jt++) {
      v16bf bb = load_fragB(qb + ((size_t)b * QLEN + jt * 16 + l16) * KD2P + kc * 32, half);
      acc[jt] = wmma_bf16(a, bb, acc[jt]);
    }
  }
  const float bcqv = bcq[0];
#pragma unroll
  for (int jt = 0; jt < 3; jt++) {
    const int j = jt * 16 + l16;
#pragma unroll
    for (int r = 0; r < 8; r++) {
      const int i = iBase + wv * 16 + half * 8 + r;
      if (i < CLEN)
        s[((size_t)b * CLEN + i) * QLEN + j] =
            acc[jt][r] + sc[b * CLEN + i] + sq[b * QLEN + j] + bcqv;
    }
  }
}

// softmax over j (48); writes row-max (reused for q2c attention).
// a written bf16, j-padded to 64 for the c2q WMMA.
__global__ __launch_bounds__(256) void softmax_j(
    const float* __restrict__ s, bf16* __restrict__ aBf,
    float* __restrict__ smax, int N) {
  const int row = blockIdx.x * 8 + (threadIdx.x >> 5);
  const int lane = threadIdx.x & 31;
  if (row >= N) return;
  const float* sp = s + (size_t)row * QLEN;
  const float v0 = sp[lane];
  const float v1 = (lane < 16) ? sp[32 + lane] : -1e30f;
  float m = fmaxf(v0, v1);
  for (int off = 16; off; off >>= 1) m = fmaxf(m, __shfl_down(m, off));
  m = __shfl(m, 0);
  const float e0 = __expf(v0 - m);
  const float e1 = (lane < 16) ? __expf(v1 - m) : 0.0f;
  float sum = e0 + e1;
  for (int off = 16; off; off >>= 1) sum += __shfl_down(sum, off);
  sum = __shfl(sum, 0);
  const float inv = 1.0f / sum;
  aBf[(size_t)row * 64 + lane] = (bf16)(e0 * inv);
  aBf[(size_t)row * 64 + 32 + lane] = (bf16)((lane < 16) ? e1 * inv : 0.0f);
  if (lane == 0) smax[row] = m;
}

// softmax over i (800) per batch
__global__ __launch_bounds__(256) void softmax_i(const float* __restrict__ smax,
                                                 float* __restrict__ batt) {
  __shared__ float red[8];
  const int b = blockIdx.x;
  const int tid = threadIdx.x, lane = tid & 31, wv = tid >> 5;
  const float* x = smax + b * CLEN;
  float m = -1e30f;
  for (int i = tid; i < CLEN; i += 256) m = fmaxf(m, x[i]);
  for (int off = 16; off; off >>= 1) m = fmaxf(m, __shfl_down(m, off));
  if (lane == 0) red[wv] = m;
  __syncthreads();
  if (tid == 0) {
    float mm = red[0];
    for (int i = 1; i < 8; i++) mm = fmaxf(mm, red[i]);
    red[0] = mm;
  }
  __syncthreads();
  m = red[0];
  __syncthreads();
  float s = 0.0f;
  for (int i = tid; i < CLEN; i += 256) s += __expf(x[i] - m);
  for (int off = 16; off; off >>= 1) s += __shfl_down(s, off);
  if (lane == 0) red[wv] = s;
  __syncthreads();
  if (tid == 0) {
    float ss = 0.0f;
    for (int i = 0; i < 8; i++) ss += red[i];
    red[0] = ss;
  }
  __syncthreads();
  const float inv = 1.0f / red[0];
  for (int i = tid; i < CLEN; i += 256) batt[b * CLEN + i] = __expf(x[i] - m) * inv;
}

// q2c[b,d] = sum_i batt[b,i] * cEnc[b,i,d]
__global__ void q2c_kernel(const float* __restrict__ batt,
                           const float* __restrict__ cEnc,
                           float* __restrict__ q2c) {
  const int b = blockIdx.x, d = threadIdx.x;
  if (d >= DD2) return;
  float s = 0.0f;
  for (int i = 0; i < CLEN; i++)
    s += batt[b * CLEN + i] * cEnc[((size_t)b * CLEN + i) * DD2 + d];
  q2c[b * DD2 + d] = s;
}

// qT[b][d][j] bf16, j padded to 64 (B operand of c2q WMMA)
__global__ void qT_kernel(const float* __restrict__ qEnc, bf16* __restrict__ qT) {
  const size_t i = (size_t)blockIdx.x * blockDim.x + threadIdx.x;
  if (i >= (size_t)BB * DD2 * 64) return;
  const int j = (int)(i & 63);
  const int d = (int)((i >> 6) % DD2);
  const int b = (int)(i / ((size_t)DD2 * 64));
  qT[i] = (bf16)((j < QLEN) ? qEnc[((size_t)b * QLEN + j) * DD2 + d] : 0.0f);
}

// c2q[b,i,:] = a[b,i,:] * q[b,:,:]   (WMMA, K=48 padded to 64)
__global__ __launch_bounds__(128) void c2q_wmma(
    const bf16* __restrict__ aBf, const bf16* __restrict__ qT,
    float* __restrict__ c2q) {
  const int b = blockIdx.y;
  const int iBase = blockIdx.x * 64;
  const int tid = threadIdx.x, lane = tid & 31, wv = tid >> 5;
  const int l16 = lane & 15, half = lane >> 4;
  int irow = iBase + wv * 16 + l16;
  if (irow >= CLEN) irow = CLEN - 1;
  const bf16* arow = aBf + ((size_t)b * CLEN + irow) * 64;
  v16bf a0 = load_fragA(arow, half);
  v16bf a1 = load_fragA(arow + 32, half);
  for (int mt = 0; mt < 25; mt++) {
    const int d = mt * 16 + l16;
    const bf16* brow = qT + ((size_t)b * DD2 + d) * 64;
    v8f acc = vzero8();
    acc = wmma_bf16(a0, load_fragB(brow, half), acc);
    acc = wmma_bf16(a1, load_fragB(brow + 32, half), acc);
#pragma unroll
    for (int r = 0; r < 8; r++) {
      const int i = iBase + wv * 16 + half * 8 + r;
      if (i < CLEN) c2q[((size_t)b * CLEN + i) * DD2 + d] = acc[r];
    }
  }
}

// g = [c | c2q | c∘c2q | c∘q2c]   (N, 1600)
__global__ void build_g(const float* __restrict__ c, const float* __restrict__ c2q,
                        const float* __restrict__ q2c, float* __restrict__ g) {
  const size_t i = (size_t)blockIdx.x * blockDim.x + threadIdx.x;
  if (i >= (size_t)NCR * GG8) return;
  const int d = (int)(i % GG8);
  const size_t n = i / GG8;
  const int b = (int)(n / CLEN);
  float v;
  if (d < 400) v = c[n * DD2 + d];
  else if (d < 800) v = c2q[n * DD2 + (d - 400)];
  else if (d < 1200) v = c[n * DD2 + (d - 800)] * c2q[n * DD2 + (d - 800)];
  else v = c[n * DD2 + (d - 1200)] * q2c[b * DD2 + (d - 1200)];
  g[i] = v;
}

// p[n] = g[n,:].gw + gb + m[n,:].mw + mb
__global__ __launch_bounds__(256) void p_out(
    const float* __restrict__ G, const float* __restrict__ gw,
    const float* __restrict__ gb, const float* __restrict__ Mm,
    const float* __restrict__ mw, const float* __restrict__ mb,
    float* __restrict__ outp) {
  const int row = blockIdx.x * 8 + (threadIdx.x >> 5);
  const int lane = threadIdx.x & 31;
  if (row >= NCR) return;
  float s = 0.0f;
  const float* gp = G + (size_t)row * GG8;
  for (int d = lane; d < GG8; d += 32) s += gp[d] * gw[d];
  const float* mp = Mm + (size_t)row * DD2;
  for (int d = lane; d < DD2; d += 32) s += mp[d] * mw[d];
  for (int off = 16; off; off >>= 1) s += __shfl_down(s, off);
  if (lane == 0) outp[row] = s + gb[0] + mb[0];
}

// ---------------------------------------------------------------------------
// Host orchestration
// ---------------------------------------------------------------------------
static inline char* ws_take(char*& cur, size_t bytes) {
  char* p = cur;
  cur += (bytes + 255) & ~(size_t)255;
  return p;
}

extern "C" void kernel_launch(void* const* d_in, const int* in_sizes, int n_in,
                              void* d_out, int out_size, void* d_ws, size_t ws_size,
                              hipStream_t stream) {
  (void)in_sizes; (void)n_in; (void)out_size; (void)ws_size;
  auto F = [&](int i) { return (const float*)d_in[i]; };
  auto I = [&](int i) { return (const int*)d_in[i]; };

  // input index map (setup_inputs dict order, params flattened recursively):
  // 0 c_char 1 q_char 2 c_word 3 q_word 4 word_emb 5 char_emb 6 conv_w 7 conv_b
  // 8..15 highway ; 16..23 ctx ; 24..29 att vecs ; 30..37 mod1 ; 38..45 mod2 ;
  // 46..53 out_lstm ; 54..61 output heads
  const int* c_char = I(0);
  const int* q_char = I(1);
  const int* c_word = I(2);
  const int* q_word = I(3);
  const float* word_emb = F(4);
  const float* char_emb = F(5);
  const float* conv_w = F(6);
  const float* conv_b = F(7);

  float* out = (float*)d_out;
  char* cur = (char*)d_ws;

  // ---- bf16 weight images (one-time conversion) ----
  bf16* whhBf[8];
  const int whhIdx[8] = {17, 21, 31, 35, 39, 43, 47, 51};
  for (int i = 0; i < 8; i++) whhBf[i] = (bf16*)ws_take(cur, (size_t)GG4 * KPAD * 2);
  bf16* hwBf[4];  // lin0, gate0, lin1, gate1  (400 x 416)
  const int hwIdx[4] = {8, 10, 12, 14};
  for (int i = 0; i < 4; i++) hwBf[i] = (bf16*)ws_take(cur, (size_t)DD2 * KD2P * 2);
  bf16* wihBf[8];  // ctxF,ctxB (800x416) mod1F,mod1B (800x1600) mod2F,mod2B,outF,outB (800x416)
  const int wihIdx[8] = {16, 20, 30, 34, 38, 42, 46, 50};
  const int wihK[8] = {KD2P, KD2P, GG8, GG8, KD2P, KD2P, KD2P, KD2P};
  const int wihD[8] = {DD2, DD2, GG8, GG8, DD2, DD2, DD2, DD2};
  for (int i = 0; i < 8; i++) wihBf[i] = (bf16*)ws_take(cur, (size_t)GG4 * wihK[i] * 2);

  // ---- activations / scratch ----
  float* cChar = (float*)ws_take(cur, (size_t)NCR * 100 * 4);
  float* qChar = (float*)ws_take(cur, (size_t)NQR * 100 * 4);
  float* cX    = (float*)ws_take(cur, (size_t)NCR * DD2 * 4);
  float* qX    = (float*)ws_take(cur, (size_t)NQR * DD2 * 4);
  float* tmpA  = (float*)ws_take(cur, (size_t)NCR * DD2 * 4);  // later: c2q
  float* tmpB  = (float*)ws_take(cur, (size_t)NCR * DD2 * 4);  // later: m2
  float* cEnc  = (float*)ws_take(cur, (size_t)NCR * DD2 * 4);
  float* qEnc  = (float*)ws_take(cur, (size_t)NQR * DD2 * 4);
  float* xwF   = (float*)ws_take(cur, (size_t)NCR * GG4 * 4);
  float* xwB   = (float*)ws_take(cur, (size_t)NCR * GG4 * 4);
  float* sc    = (float*)ws_take(cur, (size_t)NCR * 4);
  float* sq    = (float*)ws_take(cur, (size_t)NQR * 4);
  // shared bf16 staging: cXbf -> cw -> m1bf -> mbf (stream-ordered reuse)
  bf16* actBf  = (bf16*)ws_take(cur, (size_t)NCR * KD2P * 2);
  bf16* qActBf = (bf16*)ws_take(cur, (size_t)NQR * KD2P * 2);  // qXbf -> qb
  float* sBuf  = (float*)ws_take(cur, (size_t)NCR * QLEN * 4);
  bf16* aBf    = (bf16*)ws_take(cur, (size_t)NCR * 64 * 2);
  float* smax  = (float*)ws_take(cur, (size_t)NCR * 4);
  float* batt  = (float*)ws_take(cur, (size_t)NCR * 4);
  float* q2c   = (float*)ws_take(cur, (size_t)BB * DD2 * 4);
  bf16* qTbf   = (bf16*)ws_take(cur, (size_t)BB * DD2 * 64 * 2);
  float* gBuf  = (float*)ws_take(cur, (size_t)NCR * GG8 * 4);
  bf16* gBf    = (bf16*)ws_take(cur, (size_t)NCR * GG8 * 2);
  float* m1    = (float*)ws_take(cur, (size_t)NCR * DD2 * 4);
  float* mBuf  = (float*)ws_take(cur, (size_t)NCR * DD2 * 4);

  auto toBf = [&](const float* src, const float* scale, bf16* dst, int N, int D, int ld) {
    bf16_build<<<((size_t)N * ld + 255) / 256, 256, 0, stream>>>(src, scale, dst, N, D, ld);
  };
  auto gemm = [&](const bf16* X, const bf16* W, const float* b0, const float* b1,
                  float* Y, int N, int Kp, int M, int act) {
    dim3 g(N / 64, (M + 63) / 64);
    gemm_wmma<<<g, 128, 0, stream>>>(X, W, b0, b1, Y, Kp, M, M, act);
  };

  // 0) weight conversions
  for (int i = 0; i < 8; i++) toBf(F(whhIdx[i]), nullptr, whhBf[i], GG4, HH, KPAD);
  for (int i = 0; i < 4; i++) toBf(F(hwIdx[i]), nullptr, hwBf[i], DD2, DD2, KD2P);
  for (int i = 0; i < 8; i++) toBf(F(wihIdx[i]), nullptr, wihBf[i], GG4, wihD[i], wihK[i]);

  // 1) char CNN + concat with word embedding
  char_conv<<<NCR, 128, 0, stream>>>(c_char, char_emb, conv_w, conv_b, cChar);
  char_conv<<<NQR, 128, 0, stream>>>(q_char, char_emb, conv_w, conv_b, qChar);
  concat_cw<<<((size_t)NCR * DD2 + 255) / 256, 256, 0, stream>>>(cChar, c_word, word_emb, cX, NCR);
  concat_cw<<<((size_t)NQR * DD2 + 255) / 256, 256, 0, stream>>>(qChar, q_word, word_emb, qX, NQR);

  // 2) highway x2 (c and q share weights)
  for (int L = 0; L < 2; L++) {
    toBf(cX, nullptr, actBf, NCR, DD2, KD2P);
    gemm(actBf, hwBf[2 * L], F(9 + 4 * L), nullptr, tmpA, NCR, KD2P, DD2, 1);
    gemm(actBf, hwBf[2 * L + 1], F(11 + 4 * L), nullptr, tmpB, NCR, KD2P, DD2, 2);
    hw_combine<<<((size_t)NCR * DD2 + 255) / 256, 256, 0, stream>>>(cX, tmpA, tmpB, (size_t)NCR * DD2);
    toBf(qX, nullptr, qActBf, NQR, DD2, KD2P);
    gemm(qActBf, hwBf[2 * L], F(9 + 4 * L), nullptr, tmpA, NQR, KD2P, DD2, 1);
    gemm(qActBf, hwBf[2 * L + 1], F(11 + 4 * L), nullptr, tmpB, NQR, KD2P, DD2, 2);
    hw_combine<<<((size_t)NQR * DD2 + 255) / 256, 256, 0, stream>>>(qX, tmpA, tmpB, (size_t)NQR * DD2);
  }

  auto bilstm = [&](const bf16* Xbf, int N, int T, int Kp, int pbase,
                    const bf16* wihF, const bf16* wihB,
                    const bf16* whF, const bf16* whB, float* Yout) {
    gemm(Xbf, wihF, F(pbase + 2), F(pbase + 3), xwF, N, Kp, GG4, 0);
    gemm(Xbf, wihB, F(pbase + 6), F(pbase + 7), xwB, N, Kp, GG4, 0);
    lstm_dir<<<2, 256, 0, stream>>>(xwF, xwB, whF, whB, Yout, T, 2 * HH);
  };

  // 3) context BiLSTM
  toBf(cX, nullptr, actBf, NCR, DD2, KD2P);
  bilstm(actBf, NCR, CLEN, KD2P, 16, wihBf[0], wihBf[1], whhBf[0], whhBf[1], cEnc);
  toBf(qX, nullptr, qActBf, NQR, DD2, KD2P);
  bilstm(qActBf, NQR, QLEN, KD2P, 16, wihBf[0], wihBf[1], whhBf[0], whhBf[1], qEnc);

  // 4) attention flow
  rowdot<<<(NCR + 7) / 8, 256, 0, stream>>>(cEnc, F(24), F(25), sc, NCR, DD2);
  rowdot<<<(NQR + 7) / 8, 256, 0, stream>>>(qEnc, F(26), F(27), sq, NQR, DD2);
  toBf(cEnc, F(28), actBf, NCR, DD2, KD2P);   // c ∘ w_cq
  toBf(qEnc, nullptr, qActBf, NQR, DD2, KD2P);
  sim_wmma<<<dim3(13, BB), 128, 0, stream>>>(actBf, qActBf, sc, sq, F(29), sBuf);
  softmax_j<<<(NCR + 7) / 8, 256, 0, stream>>>(sBuf, aBf, smax, NCR);
  softmax_i<<<BB, 256, 0, stream>>>(smax, batt);
  q2c_kernel<<<BB, 512, 0, stream>>>(batt, cEnc, q2c);
  qT_kernel<<<((size_t)BB * DD2 * 64 + 255) / 256, 256, 0, stream>>>(qEnc, qTbf);
  c2q_wmma<<<dim3(13, BB), 128, 0, stream>>>(aBf, qTbf, tmpA /* = c2q */);
  build_g<<<((size_t)NCR * GG8 + 255) / 256, 256, 0, stream>>>(cEnc, tmpA, q2c, gBuf);

  // 5) modeling stack
  toBf(gBuf, nullptr, gBf, NCR, GG8, GG8);
  bilstm(gBf, NCR, CLEN, GG8, 30, wihBf[2], wihBf[3], whhBf[2], whhBf[3], m1);
  toBf(m1, nullptr, actBf, NCR, DD2, KD2P);
  bilstm(actBf, NCR, CLEN, KD2P, 38, wihBf[4], wihBf[5], whhBf[4], whhBf[5], mBuf);

  // 6) p1
  p_out<<<(NCR + 7) / 8, 256, 0, stream>>>(gBuf, F(54), F(55), mBuf, F(56), F(57), out);

  // 7) output BiLSTM + p2
  toBf(mBuf, nullptr, actBf, NCR, DD2, KD2P);
  bilstm(actBf, NCR, CLEN, KD2P, 46, wihBf[6], wihBf[7], whhBf[6], whhBf[7], tmpB /* m2 */);
  p_out<<<(NCR + 7) / 8, 256, 0, stream>>>(gBuf, F(58), F(59), tmpB, F(60), F(61), out + NCR);
}
